// GatedGraphConvolution_80547816669790
// MI455X (gfx1250) — compile-verified, matrix operand order checked by reading, not run on hardware
//
#include <hip/hip_runtime.h>
#include <hip/hip_bf16.h>

// ---------------- problem constants ----------------
#define FDIM   128      // node feature dim == out features
#define INF    384      // 3*F
#define K1DIM  128
#define K2DIM  26       // padded to 32 for WMMA
#define NT     8        // N tiles of 16 (128/16)
#define KS_G   12       // 384/32
#define KS_1   4        // 128/32

// packed bf16 weight workspace layout (element offsets)
#define SZ_G   (NT*KS_G*512)   // 49152
#define SZ_1   (NT*KS_1*512)   // 16384
#define SZ_2   (NT*512)        // 4096
#define OFF_G  0
#define OFF_M  (OFF_G + SZ_G)          // 49152
#define OFF_1  (OFF_M + SZ_G)          // 98304
#define OFF_2  (OFF_1 + SZ_1)          // 114688
#define OFF_2G (OFF_2 + SZ_2)          // 118784
#define PACK_TOTAL (OFF_2G + SZ_2)     // 122880 bf16 elements (240 KB)

// LDS row strides (ushorts), padded so rows land on distinct bank groups
#define STRA 392   // 384 + 8
#define STRC 136   // 128 + 8
#define STRP 40    //  32 + 8

typedef __attribute__((ext_vector_type(16))) __bf16    v16bf;
typedef __attribute__((ext_vector_type(8)))  float     v8f;
typedef __attribute__((ext_vector_type(4)))  float     f32x4;
typedef __attribute__((ext_vector_type(4)))  unsigned  u32x4;
typedef __attribute__((ext_vector_type(2)))  unsigned  u32x2;

union FragBF { u32x4 u[2]; v16bf v; };

// float -> bf16 through the native convert path
__device__ __forceinline__ unsigned short f2bf(float f) {
    union { __bf16 h; unsigned short s; } u;
    u.h = (__bf16)f;
    return u.s;
}
// pack two floats into one DWORD of bf16x2 (lets clang use v_cvt_pk_bf16_f32)
__device__ __forceinline__ unsigned pk2(float lo, float hi) {
    return (unsigned)f2bf(lo) | ((unsigned)f2bf(hi) << 16);
}

__device__ __forceinline__ float sigmoidf(float x) {
    return 1.0f / (1.0f + __expf(-x));
}

// A fragment (16x32 bf16) from LDS, row-major with padded stride.
// 16-bit A layout: lanes 0-15 -> K 0..7 & 16..23, lanes 16-31 -> K 8..15 & 24..31.
__device__ __forceinline__ v16bf load_a16(const unsigned short* base, int stride,
                                          int ks, int lane) {
    const int mm = lane & 15, half = lane >> 4;
    const unsigned short* p = base + mm * stride + ks * 32 + half * 8;
    FragBF f;
    f.u[0] = *(const u32x4*)(p);        // e0..7  -> K base + 0..7
    f.u[1] = *(const u32x4*)(p + 16);   // e8..15 -> K base + 16..23
    return f.v;
}

// B fragment: pre-packed lane-major, 32 bytes contiguous per lane.
__device__ __forceinline__ v16bf load_bfrag(const unsigned short* tile, int lane) {
    const unsigned short* p = tile + lane * 16;
    FragBF f;
    f.u[0] = *(const u32x4*)(p);
    f.u[1] = *(const u32x4*)(p + 8);
    return f.v;
}

// ---------------- weight packing: fp32 [out,in] -> bf16 B-fragments ----------------
__global__ void pack_weights_kernel(const float* __restrict__ Wg,
                                    const float* __restrict__ Wm,
                                    const float* __restrict__ W1,
                                    const float* __restrict__ W2,
                                    const float* __restrict__ W2g,
                                    unsigned short* __restrict__ out) {
    int idx = blockIdx.x * 256 + threadIdx.x;
    if (idx >= PACK_TOTAL) return;

    const float* W; int ksteps, Kdim; int rem = idx;
    if (rem < SZ_G)                { W = Wg;  ksteps = KS_G; Kdim = INF;   }
    else if ((rem -= SZ_G) < SZ_G) { W = Wm;  ksteps = KS_G; Kdim = INF;   }
    else if ((rem -= SZ_G) < SZ_1) { W = W1;  ksteps = KS_1; Kdim = K1DIM; }
    else if ((rem -= SZ_1) < SZ_2) { W = W2;  ksteps = 1;    Kdim = K2DIM; }
    else { rem -= SZ_2;              W = W2g; ksteps = 1;    Kdim = K2DIM; }

    const int tile  = rem >> 9;          // /512
    const int lane  = (rem >> 4) & 31;
    const int e     = rem & 15;
    const int ntile = tile / ksteps;
    const int ks    = tile % ksteps;
    const int n     = ntile * 16 + (lane & 15);
    const int half  = lane >> 4;
    const int k     = ks * 32 + ((e < 8) ? (half * 8 + e)
                                         : (16 + half * 8 + (e - 8)));
    const float v = (k < Kdim) ? W[(size_t)n * Kdim + k] : 0.0f;
    out[idx] = f2bf(v);
}

// ---------------- fused edge kernel ----------------
__global__ __launch_bounds__(256)
void edge_kernel(const float* __restrict__ input,
                 const int*   __restrict__ esrc,
                 const int*   __restrict__ etgt,
                 const float* __restrict__ rij,
                 const float* __restrict__ combine,
                 const float* __restrict__ pw,
                 const unsigned short* __restrict__ packed,
                 float* __restrict__ out, int E) {
    __shared__ __align__(16) unsigned short sA[16 * STRA];  // final_fe bf16
    __shared__ __align__(16) unsigned short sC[16 * STRC];  // combine tile bf16
    __shared__ __align__(16) unsigned short sP[16 * STRP];  // plane_wave tile (padded)
    __shared__ int sSrc[16];

    const int tid  = threadIdx.x;
    const int tile = blockIdx.x;

    // -------- phase 1: gather + stage (bf16 convert, b128 LDS stores) --------
    {
        const int m = tid >> 4, c = tid & 15;          // edge-in-tile, 8-col chunk
        int e = tile * 16 + m; if (e >= E) e = E - 1;
        if (tid < 16) {
            int ee = tile * 16 + tid; if (ee >= E) ee = E - 1;
            sSrc[tid] = esrc[ee];
        }
        const int   src = esrc[e], tgt = etgt[e];
        const float inv = 1.0f / rij[e];

        const f32x4* pi = (const f32x4*)(input + (size_t)src * FDIM + c * 8);
        const f32x4* pj = (const f32x4*)(input + (size_t)tgt * FDIM + c * 8);
        f32x4 a0 = pi[0], a1 = pi[1];
        f32x4 b0 = pj[0], b1 = pj[1];
        float ai[8] = {a0.x, a0.y, a0.z, a0.w, a1.x, a1.y, a1.z, a1.w};
        float bj[8] = {b0.x, b0.y, b0.z, b0.w, b1.x, b1.y, b1.z, b1.w};

        unsigned short* row = sA + m * STRA;
        u32x4 wi = {pk2(ai[0], ai[1]), pk2(ai[2], ai[3]),
                    pk2(ai[4], ai[5]), pk2(ai[6], ai[7])};
        u32x4 wj = {pk2(bj[0], bj[1]), pk2(bj[2], bj[3]),
                    pk2(bj[4], bj[5]), pk2(bj[6], bj[7])};
        float d[8];
        #pragma unroll
        for (int i = 0; i < 8; i++) d[i] = (bj[i] - ai[i]) * inv;
        u32x4 wd = {pk2(d[0], d[1]), pk2(d[2], d[3]),
                    pk2(d[4], d[5]), pk2(d[6], d[7])};
        *(u32x4*)(row + c * 8)       = wi;
        *(u32x4*)(row + 128 + c * 8) = wj;
        *(u32x4*)(row + 256 + c * 8) = wd;

        // combine_sets streams once: non-temporal to protect L2 residency
        const f32x4* pc = (const f32x4*)(combine + (size_t)e * K1DIM + c * 8);
        f32x4 c0 = __builtin_nontemporal_load(pc);
        f32x4 c1 = __builtin_nontemporal_load(pc + 1);
        u32x4 wc = {pk2(c0.x, c0.y), pk2(c0.z, c0.w),
                    pk2(c1.x, c1.y), pk2(c1.z, c1.w)};
        *(u32x4*)(sC + m * STRC + c * 8) = wc;
    }
    if (tid < 128) {                                   // plane_wave: 16 x 32 (pad 26->32)
        const int m2 = tid >> 3, c2 = tid & 7;
        int e2 = tile * 16 + m2; if (e2 >= E) e2 = E - 1;
        float v[4];
        #pragma unroll
        for (int i = 0; i < 4; i++) {
            const int j = c2 * 4 + i;
            v[i] = (j < K2DIM)
                 ? __builtin_nontemporal_load(pw + (size_t)e2 * K2DIM + j) : 0.0f;
        }
        u32x2 wv = {pk2(v[0], v[1]), pk2(v[2], v[3])};
        *(u32x2*)(sP + m2 * STRP + c2 * 4) = wv;
    }
    __syncthreads();

    // -------- phase 2: per-wave 16x16 WMMA tiles --------
    const int w    = tid >> 5;      // wave id == N tile (0..7)
    const int lane = tid & 31;

    v8f accG = {}, accM = {}, acc1 = {}, acc2 = {}, acc2g = {};

    #pragma unroll
    for (int ks = 0; ks < KS_G; ks++) {     // e_gate + e_MLP share the A fragment
        v16bf a  = load_a16(sA, STRA, ks, lane);
        v16bf bg = load_bfrag(packed + OFF_G + (size_t)(w * KS_G + ks) * 512, lane);
        v16bf bm = load_bfrag(packed + OFF_M + (size_t)(w * KS_G + ks) * 512, lane);
        accG = __builtin_amdgcn_wmma_f32_16x16x32_bf16(false, a, false, bg,
                                                       (short)0, accG, false, false);
        accM = __builtin_amdgcn_wmma_f32_16x16x32_bf16(false, a, false, bm,
                                                       (short)0, accM, false, false);
    }
    #pragma unroll
    for (int ks = 0; ks < KS_1; ks++) {     // z1
        v16bf a = load_a16(sC, STRC, ks, lane);
        v16bf b = load_bfrag(packed + OFF_1 + (size_t)(w * KS_1 + ks) * 512, lane);
        acc1 = __builtin_amdgcn_wmma_f32_16x16x32_bf16(false, a, false, b,
                                                       (short)0, acc1, false, false);
    }
    {                                        // z2 (value + gate) share the A fragment
        v16bf a  = load_a16(sP, STRP, 0, lane);
        v16bf b2 = load_bfrag(packed + OFF_2  + (size_t)w * 512, lane);
        v16bf bg = load_bfrag(packed + OFF_2G + (size_t)w * 512, lane);
        acc2  = __builtin_amdgcn_wmma_f32_16x16x32_bf16(false, a, false, b2,
                                                        (short)0, acc2, false, false);
        acc2g = __builtin_amdgcn_wmma_f32_16x16x32_bf16(false, a, false, bg,
                                                        (short)0, acc2g, false, false);
    }

    // -------- epilogue: activations + index_add scatter --------
    const int half = lane >> 4;
    const int n    = w * 16 + (lane & 15);
    #pragma unroll
    for (int r = 0; r < 8; r++) {
        const int mm   = r + half * 8;                 // C/D layout row
        const int eidx = tile * 16 + mm;
        if (eidx < E) {
            float g  = sigmoidf(accG[r]);
            float ml = accM[r];
            ml = (ml > 0.0f) ? ml : (__expf(ml) - 1.0f);   // ELU
            float z2 = acc2[r] * sigmoidf(acc2g[r]);
            float z  = g * ml * (acc1[r] + z2);
            atomicAdd(out + (size_t)sSrc[mm] * FDIM + n, z);
        }
    }
}

// ---------------- host-side launch ----------------
extern "C" void kernel_launch(void* const* d_in, const int* in_sizes, int n_in,
                              void* d_out, int out_size, void* d_ws, size_t ws_size,
                              hipStream_t stream) {
    const float* input   = (const float*)d_in[0];
    // d_in[1] = nodes (unused)
    const int*   esrc    = (const int*)d_in[2];
    const int*   etgt    = (const int*)d_in[3];
    const float* rij     = (const float*)d_in[4];
    const float* combine = (const float*)d_in[5];
    const float* pw      = (const float*)d_in[6];
    const float* W1      = (const float*)d_in[7];
    const float* W2      = (const float*)d_in[8];
    const float* W2g     = (const float*)d_in[9];
    const float* Wg      = (const float*)d_in[10];
    const float* Wm      = (const float*)d_in[11];

    const int E = in_sizes[2];
    unsigned short* packed = (unsigned short*)d_ws;   // needs 240 KB

    // output = input.clone()
    (void)hipMemcpyAsync(d_out, d_in[0], (size_t)in_sizes[0] * sizeof(float),
                         hipMemcpyDeviceToDevice, stream);

    pack_weights_kernel<<<(PACK_TOTAL + 255) / 256, 256, 0, stream>>>(
        Wg, Wm, W1, W2, W2g, packed);

    const int tiles = (E + 15) / 16;
    edge_kernel<<<tiles, 256, 0, stream>>>(input, esrc, etgt, rij, combine, pw,
                                           packed, (float*)d_out, E);
}
